// GNN_9062380995258
// MI455X (gfx1250) — compile-verified, matrix operand order checked by reading, not run on hardware
//
#include <hip/hip_runtime.h>
#include <hip/hip_bf16.h>
#include <stdint.h>

#define NN   50000
#define NE   320000
#define HID  256

typedef __attribute__((ext_vector_type(16))) __bf16          v16bf;
typedef __attribute__((ext_vector_type(16))) unsigned short  v16u16;
typedef __attribute__((ext_vector_type(8)))  float           v8f;

// Native bf16 convert (RTNE) -> v_cvt_bf16_f32 / v_cvt_pk_bf16_f32
static __device__ __forceinline__ unsigned short f32_to_bf16_bits(float f) {
  return __builtin_bit_cast(unsigned short, (__bf16)f);
}

// ---------------- utility kernels ----------------

__global__ void zero_f32(float* p, int n4) {      // n4 = number of float4's
  int i = blockIdx.x * blockDim.x + threadIdx.x;
  int stride = gridDim.x * blockDim.x;
  float4 z = make_float4(0.f, 0.f, 0.f, 0.f);
  for (; i < n4; i += stride) ((float4*)p)[i] = z;
}

// Wc = eW2 @ nW1[256:512,:]   (f32, tiny: 16 MFLOP)
__global__ void wc_gemm(const float* __restrict__ eW2,
                        const float* __restrict__ nW1,
                        float* __restrict__ wc) {
  int n = threadIdx.x, k = blockIdx.x;
  float acc = 0.f;
  for (int m = 0; m < HID; ++m)
    acc = fmaf(eW2[k * HID + m], nW1[(HID + m) * HID + n], acc);
  wc[k * HID + n] = acc;
}

// Pack a 256x256 f32 matrix into bf16 WMMA B-fragments (16x16x32 bf16 shape).
// Fragment (kt,nt): lane's 16 values contiguous at lane*16 (32B -> b128 loads).
// K per lane follows the documented 16-bit A/B striping:
//   lane<16 : K = {0..7, 16..23};  lane>=16 : K = {8..15, 24..31}
struct FragSrc { const float* src[8]; };

__global__ void pack_frags(FragSrc fs, unsigned short* __restrict__ fragbuf) {
  int m    = blockIdx.x >> 7;        // matrix 0..7
  int f    = blockIdx.x & 127;       // fragment 0..127
  int kt   = f >> 4, nt = f & 15;
  int lane = threadIdx.x >> 4;       // 0..31  (512 threads)
  int e    = threadIdx.x & 15;       // 0..15
  int half = e >> 3, within = e & 7;
  int koff = half * 16 + within + ((lane < 16) ? 0 : 8);
  int k = kt * 32 + koff;
  int n = nt * 16 + (lane & 15);
  float v = fs.src[m][k * HID + n];
  fragbuf[(size_t)m * 65536 + (size_t)f * 512 + lane * 16 + e] = f32_to_bf16_bits(v);
}

// ---------------- edge scatter kernel ----------------
// Per edge e: u = relu(a_e*eW1 + eb1);  U[i] += u;  S[i] += x[j] + eb2
// One wave per edge, 8 channels per lane. Weights cached in registers.
__global__ void edge_kernel(const float* __restrict__ x,
                            const int*   __restrict__ eidx,
                            const float* __restrict__ eattr,
                            const float* __restrict__ eW1,
                            const float* __restrict__ eb1,
                            const float* __restrict__ eb2,
                            float* __restrict__ U, float* __restrict__ S) {
  int lane   = threadIdx.x & 31;
  int wave   = blockIdx.x * (blockDim.x >> 5) + (threadIdx.x >> 5);
  int nwaves = gridDim.x * (blockDim.x >> 5);
  int c0     = lane * 8;

  float w1[8], b1[8], b2[8];
#pragma unroll
  for (int t = 0; t < 8; ++t) { w1[t] = eW1[c0+t]; b1[t] = eb1[c0+t]; b2[t] = eb2[c0+t]; }

  for (int e = wave; e < NE; e += nwaves) {
    float a = eattr[e];
    int i = eidx[e], j = eidx[NE + e];
    const float* xr = x + (size_t)j * HID + c0;
    float* Ur = U + (size_t)i * HID + c0;
    float* Sr = S + (size_t)i * HID + c0;
    float4 xa = *(const float4*)(xr);
    float4 xb = *(const float4*)(xr + 4);
    float xs[8] = { xa.x, xa.y, xa.z, xa.w, xb.x, xb.y, xb.z, xb.w };
#pragma unroll
    for (int t = 0; t < 8; ++t) {
      float u = fmaxf(fmaf(a, w1[t], b1[t]), 0.f);
      atomicAdd(Ur + t, u);
      atomicAdd(Sr + t, xs[t] + b2[t]);
    }
  }
}

// ---------------- WMMA fragment helpers ----------------

static __device__ __forceinline__ v16bf pack8x2(const float* p0, const float* p1) {
  v16bf t;
#pragma unroll
  for (int q = 0; q < 8; ++q) t[q]     = (__bf16)p0[q];
#pragma unroll
  for (int q = 0; q < 8; ++q) t[8 + q] = (__bf16)p1[q];
  return t;
}

// A fragment (16x32, M x K) from row-major f32 global matrix.
static __device__ __forceinline__ v16bf a_frag_global(const float* base, int row,
                                                      int k0, int lane) {
  const float* p = base + (size_t)row * HID + k0 + ((lane >> 4) << 3);
  return pack8x2(p, p + 16);
}

// A fragment from bf16 LDS tile with padded stride 264 (bank-conflict free).
static __device__ __forceinline__ v16bf a_frag_lds(const unsigned short* t, int row,
                                                   int k0, int lane) {
  const unsigned short* p = t + row * 264 + k0 + ((lane >> 4) << 3);
  v16u16 r;
#pragma unroll
  for (int q = 0; q < 8; ++q) r[q]     = p[q];
#pragma unroll
  for (int q = 0; q < 8; ++q) r[8 + q] = p[16 + q];
  return __builtin_bit_cast(v16bf, r);
}

// B fragment from pre-packed bf16 weights.
static __device__ __forceinline__ v16bf b_frag(const unsigned short* w, int kt,
                                               int nt, int lane) {
  const v16u16* p = (const v16u16*)(w + (size_t)(kt * 16 + nt) * 512 + lane * 16);
  return __builtin_bit_cast(v16bf, *p);
}

// ---------------- fused node-update kernel ----------------
// out[m] = relu( x@W1top + U@Wc + S@W1bot + nb1 ) @ W2 + nb2 + x[m]
// Workgroup: 64 rows x 256 cols. 8 waves, each 16 rows x 128 cols
// (8 accumulators of 16x16 f32).
__global__ void __launch_bounds__(256) node_kernel(
    const float* x, const float* __restrict__ U, const float* __restrict__ S,
    const unsigned short* __restrict__ fW1top, const unsigned short* __restrict__ fWc,
    const unsigned short* __restrict__ fW1bot, const unsigned short* __restrict__ fW2,
    const float* __restrict__ nb1, const float* __restrict__ nb2,
    float* out) {
  __shared__ unsigned short tbuf[64 * 264];   // bf16 activations, padded stride

  int lane    = threadIdx.x & 31;
  int w       = threadIdx.x >> 5;     // wave 0..7
  int msub    = (w & 3) * 16;         // row sub-tile
  int nhalf   = w >> 2;               // column half (0 or 1)
  int rowbase = blockIdx.x * 64;

  int arow_local = msub + (lane & 15);
  int arow = rowbase + arow_local;
  if (arow > NN - 1) arow = NN - 1;   // clamp reads; stores are masked

  v8f acc[8] = {};

  // Stage 1: three K=256 GEMM accumulations -> 16x128 per wave
#pragma unroll
  for (int s = 0; s < 3; ++s) {
    const float* src = (s == 0) ? x : (s == 1) ? U : S;
    const unsigned short* wt = (s == 0) ? fW1top : (s == 1) ? fWc : fW1bot;
    for (int kt = 0; kt < 8; ++kt) {
      v16bf a = a_frag_global(src, arow, kt * 32, lane);
#pragma unroll
      for (int q = 0; q < 8; ++q) {
        v16bf b = b_frag(wt, kt, nhalf * 8 + q, lane);
        acc[q] = __builtin_amdgcn_wmma_f32_16x16x32_bf16(
            false, a, false, b, (short)0, acc[q], false, false);
      }
    }
  }

  // Epilogue 1: bias + relu, convert to bf16, stash in LDS
  int roff = (lane >> 4) * 8;
#pragma unroll
  for (int q = 0; q < 8; ++q) {
    int col = nhalf * 128 + q * 16 + (lane & 15);
    float b = nb1[col];
#pragma unroll
    for (int r = 0; r < 8; ++r) {
      int row = msub + r + roff;
      tbuf[row * 264 + col] = f32_to_bf16_bits(fmaxf(acc[q][r] + b, 0.f));
    }
  }
  __syncthreads();

  // Stage 2: t @ nW2
  v8f acc2[8] = {};
  for (int kt = 0; kt < 8; ++kt) {
    v16bf a = a_frag_lds(tbuf, arow_local, kt * 32, lane);
#pragma unroll
    for (int q = 0; q < 8; ++q) {
      v16bf b = b_frag(fW2, kt, nhalf * 8 + q, lane);
      acc2[q] = __builtin_amdgcn_wmma_f32_16x16x32_bf16(
          false, a, false, b, (short)0, acc2[q], false, false);
    }
  }

  // Epilogue 2: + nb2 + residual x, masked store
#pragma unroll
  for (int q = 0; q < 8; ++q) {
    int col = nhalf * 128 + q * 16 + (lane & 15);
    float b = nb2[col];
#pragma unroll
    for (int r = 0; r < 8; ++r) {
      int row = rowbase + msub + r + roff;
      if (row < NN)
        out[(size_t)row * HID + col] = acc2[q][r] + b + x[(size_t)row * HID + col];
    }
  }
}

// ---------------- host launcher ----------------

extern "C" void kernel_launch(void* const* d_in, const int* in_sizes, int n_in,
                              void* d_out, int out_size, void* d_ws, size_t ws_size,
                              hipStream_t stream) {
  (void)in_sizes; (void)n_in; (void)out_size; (void)ws_size;
  const float* z     = (const float*)d_in[0];
  const int*   eidx  = (const int*)d_in[1];
  const float* eattr = (const float*)d_in[2];
  const float* p[16];
  for (int i = 0; i < 16; ++i) p[i] = (const float*)d_in[3 + i];
  // per layer: [0]=eW1 [1]=eb1 [2]=eW2 [3]=eb2 [4]=nW1 [5]=nb1 [6]=nW2 [7]=nb2

  float* out = (float*)d_out;
  const size_t NH = (size_t)NN * HID;            // 12.8M floats
  float* U  = (float*)d_ws;
  float* S  = U + NH;
  float* wc = S + NH;                            // 2 * 65536 f32
  unsigned short* frags = (unsigned short*)(wc + 2 * 65536);  // 8 * 65536 bf16

  // Precompute Wc = eW2 @ nW1[256:512] per layer
  wc_gemm<<<256, 256, 0, stream>>>(p[2],  p[4],  wc);
  wc_gemm<<<256, 256, 0, stream>>>(p[10], p[12], wc + 65536);

  // Pack all 8 weight matrices into bf16 WMMA fragments
  FragSrc fs;
  fs.src[0] = p[4];             // L0 W1top  (nW1 rows 0..255)
  fs.src[1] = wc;               // L0 Wc
  fs.src[2] = p[4] + 65536;     // L0 W1bot  (nW1 rows 256..511)
  fs.src[3] = p[6];             // L0 nW2
  fs.src[4] = p[12];            // L1 W1top
  fs.src[5] = wc + 65536;       // L1 Wc
  fs.src[6] = p[12] + 65536;    // L1 W1bot
  fs.src[7] = p[14];            // L1 nW2
  pack_frags<<<1024, 512, 0, stream>>>(fs, frags);

  const float* xin = z;
  for (int l = 0; l < 2; ++l) {
    const float* const* q = p + l * 8;
    zero_f32<<<2048, 256, 0, stream>>>(U, (int)((2 * NH) / 4));
    edge_kernel<<<1024, 256, 0, stream>>>(xin, eidx, eattr, q[0], q[1], q[3], U, S);
    const unsigned short* fb = frags + (size_t)l * 4 * 65536;
    node_kernel<<<(NN + 63) / 64, 256, 0, stream>>>(
        xin, U, S,
        fb + 0 * 65536ul, fb + 1 * 65536ul, fb + 2 * 65536ul, fb + 3 * 65536ul,
        q[5], q[7], out);
    xin = out;   // layer 1 reads layer-0 output (row-local, hazard free)
  }
}
